// Set2Set_24953759990451
// MI455X (gfx1250) — matrix operand order repaired and checked by reading, not compiled, not run on hardware
//
#include <hip/hip_runtime.h>
#include <hip/hip_bf16.h>

// ---------------------------------------------------------------------------
// Set2Set: D=512, HID=1024, B=64, N=512 sequential steps.
// bf16 WMMA (v_wmma_f32_16x16x32_bf16) for the recurrent GEMM with manually
// unrolled-by-2, statically-named double buffers (keeps fragments in VGPRs,
// overlaps next-step loads with current WMMAs).  VALU + LDS softmax attention.
// ---------------------------------------------------------------------------

typedef __attribute__((ext_vector_type(16))) __bf16 v16bf;
typedef __attribute__((ext_vector_type(8)))  float  v8f;

#define DD    512
#define HID   1024
#define BB    64
#define NN    512
#define KTOT  1536            // HID + D  (q_star || h)
#define GOUT  2048            // 4*D gate outputs

// f32 -> bf16 round-to-nearest-even
static __device__ __forceinline__ unsigned short f2bf(float f) {
    union { float f; unsigned int u; } v; v.f = f;
    unsigned int u = v.u;
    unsigned int r = u + 0x7FFFu + ((u >> 16) & 1u);
    return (unsigned short)(r >> 16);
}

static __device__ __forceinline__ float sigmoidf_(float x) {
    return 1.0f / (1.0f + __expf(-x));
}

union FragBF { v16bf v; unsigned int u[8]; };

// A fragment (16x32 bf16): lane = row m, K = k0 + 16*g + 2*v (+{0,1})
static __device__ __forceinline__ void load_a(FragBF& a,
                                              const unsigned short* __restrict__ Arow,
                                              int g, int k0) {
#pragma unroll
    for (int v = 0; v < 8; ++v) {
        int K = k0 + 16 * g + 2 * v;
        a.u[v] = *(const unsigned int*)(Arow + K);
    }
}

// 4 gate B fragments (32x16 bf16): lane = col n, K = k0 + 16*g + 2*v (+{0,1})
static __device__ __forceinline__ void load_b4(FragBF* b,
                                               const unsigned short* const* Brow,
                                               int k0) {
#pragma unroll
    for (int q = 0; q < 4; ++q)
#pragma unroll
        for (int v = 0; v < 8; ++v)
            b[q].u[v] = *(const unsigned int*)(Brow[q] + k0 + 2 * v);
}

static __device__ __forceinline__ void wmma4(v8f* acc, const FragBF& a,
                                             const FragBF* b) {
#pragma unroll
    for (int q = 0; q < 4; ++q)
        acc[q] = __builtin_amdgcn_wmma_f32_16x16x32_bf16(
            false, a.v, false, b[q].v, (short)0, acc[q], false, false);
}

// ---------------------------------------------------------------------------
// Weight conversion: Wbf[j][k] (bf16, [2048][1536]) = [w_ih | w_hh] row j.
// ---------------------------------------------------------------------------
__global__ void convert_w_kernel(const float* __restrict__ w_ih,
                                 const float* __restrict__ w_hh,
                                 unsigned short* __restrict__ Wbf) {
    int idx = blockIdx.x * blockDim.x + threadIdx.x;
    if (idx >= GOUT * KTOT) return;
    int j = idx / KTOT;
    int k = idx - j * KTOT;
    float v = (k < HID) ? w_ih[j * HID + k] : w_hh[j * DD + (k - HID)];
    Wbf[idx] = f2bf(v);
}

__global__ void zero_kernel(unsigned int* __restrict__ p, int n) {
    int idx = blockIdx.x * blockDim.x + threadIdx.x;
    if (idx < n) p[idx] = 0u;
}

// ---------------------------------------------------------------------------
// Gates + LSTM cell.  One wave per (16-batch x 16-d) tile; the wave holds the
// i,f,g,o accumulator tiles (n = q*512 + d), runs K=1536 of bf16 WMMA with a
// 2-deep software pipeline, then fuses the LSTM cell update in-register.
// grid: 32 blocks x 128 threads = 128 waves = 4 mtiles * 32 dtiles.
// ---------------------------------------------------------------------------
__global__ __launch_bounds__(128) void gates_kernel(
    const unsigned short* __restrict__ Xin,   // [64][1536] bf16
    const unsigned short* __restrict__ Wbf,   // [2048][1536] bf16
    const float* __restrict__ b_ih,
    const float* __restrict__ b_hh,
    float* __restrict__ cstate,               // [64][512]
    float* __restrict__ qstar,                // [64][1024] f32
    unsigned short* __restrict__ Xout)        // [64][1536] bf16
{
    int wid  = (blockIdx.x * blockDim.x + threadIdx.x) >> 5;
    int lane = threadIdx.x & 31;
    int mt   = wid >> 5;          // 0..3   batch tile
    int dt   = wid & 31;          // 0..31  d tile
    int m0   = mt * 16;
    int d0   = dt * 16;
    int g    = lane >> 4;         // lane group (0/1)
    int ln   = lane & 15;

    v8f zero = {};
    v8f acc[4] = {zero, zero, zero, zero};

    const unsigned short* Arow = Xin + (m0 + ln) * KTOT;
    const unsigned short* Brow[4];
#pragma unroll
    for (int q = 0; q < 4; ++q)
        Brow[q] = Wbf + (q * DD + d0 + ln) * KTOT + 16 * g;

    // statically-named double buffers (register-resident)
    FragBF a0, a1;
    FragBF b0[4], b1[4];

    // ---- prologue: fragments for k = 0
    load_a(a0, Arow, g, 0);
    load_b4(b0, Brow, 0);

    // KTOT/32 = 48 k-steps, unrolled by 2 -> 24 iterations
    for (int k0 = 0; k0 < KTOT; k0 += 64) {
        // loads for k0+32 (always in range: KTOT is a multiple of 64)
        load_a(a1, Arow, g, k0 + 32);
        load_b4(b1, Brow, k0 + 32);
        __builtin_prefetch(Arow + k0 + 128, 0, 1);  // global_prefetch_b8

        wmma4(acc, a0, b0);                         // consumes resident regs

        // loads for k0+64 (wrap to 0 on last iteration; loaded, never used)
        int kn = (k0 + 64 < KTOT) ? (k0 + 64) : 0;
        load_a(a0, Arow, g, kn);
        load_b4(b0, Brow, kn);

        wmma4(acc, a1, b1);
    }

    // ---- fused LSTM cell. C/D layout: VGPR r -> M = r + 8*g, N = lane&15.
    int d = d0 + ln;
    float bias[4];
#pragma unroll
    for (int q = 0; q < 4; ++q)
        bias[q] = b_ih[q * DD + d] + b_hh[q * DD + d];

#pragma unroll
    for (int r = 0; r < 8; ++r) {
        int b  = m0 + 8 * g + r;
        float gi = sigmoidf_(acc[0][r] + bias[0]);
        float gf = sigmoidf_(acc[1][r] + bias[1]);
        float gg = tanhf   (acc[2][r] + bias[2]);
        float go = sigmoidf_(acc[3][r] + bias[3]);
        float cn = gf * cstate[b * DD + d] + gi * gg;
        cstate[b * DD + d] = cn;
        float h = go * tanhf(cn);
        qstar[b * HID + d] = h;
        unsigned short hb = f2bf(h);
        Xout[b * KTOT + d]       = hb;   // q_star[:, 0:512] = h
        Xout[b * KTOT + HID + d] = hb;   // LSTM h input slot
    }
}

// ---------------------------------------------------------------------------
// Attention: e = E[b] @ h, softmax over n, r = a^T E[b].
// One block per batch (512 threads, 16 waves).  h and softmax weights in LDS.
// Writes r to qstar_f32[:, 512:1024] and bf16 r into Xout cols [512,1024).
// ---------------------------------------------------------------------------
__global__ __launch_bounds__(512) void attn_kernel(
    const float* __restrict__ emb,     // [64][512][512] f32
    float* __restrict__ qstar,         // [64][1024] f32
    unsigned short* __restrict__ Xout) // [64][1536] bf16
{
    __shared__ float hs[DD];
    __shared__ float as[NN];
    __shared__ float red[17];

    int b   = blockIdx.x;
    int tid = threadIdx.x;
    int w   = tid >> 5;
    int l   = tid & 31;

    hs[tid] = qstar[b * HID + tid];    // h lives in cols 0..511
    __syncthreads();

    const float* Eb  = emb + (size_t)b * NN * DD;
    const float* row = Eb + (size_t)tid * DD;

    // e[b, tid] = E[b, tid, :] . h
    float e = 0.f;
    for (int dd = 0; dd < DD; dd += 4) {
        float4 v = *(const float4*)(row + dd);
        e += v.x * hs[dd] + v.y * hs[dd + 1] + v.z * hs[dd + 2] + v.w * hs[dd + 3];
    }

    // block max
    float m = e;
#pragma unroll
    for (int off = 16; off > 0; off >>= 1) m = fmaxf(m, __shfl_xor(m, off, 32));
    if (l == 0) red[w] = m;
    __syncthreads();
    if (tid == 0) {
        float t = red[0];
        for (int i = 1; i < 16; ++i) t = fmaxf(t, red[i]);
        red[16] = t;
    }
    __syncthreads();
    m = red[16];

    float p = __expf(e - m);
    float s = p;
#pragma unroll
    for (int off = 16; off > 0; off >>= 1) s += __shfl_xor(s, off, 32);
    __syncthreads();                 // done reading red[16]
    if (l == 0) red[w] = s;
    __syncthreads();
    if (tid == 0) {
        float t = 0.f;
        for (int i = 0; i < 16; ++i) t += red[i];
        red[16] = t;
    }
    __syncthreads();
    as[tid] = p / red[16];
    __syncthreads();

    // r[b, tid] = sum_n a[n] * E[b, n, tid]  (coalesced over tid)
    float r = 0.f;
    for (int n = 0; n < NN; ++n)
        r += as[n] * Eb[(size_t)n * DD + tid];

    qstar[b * HID + DD + tid] = r;
    Xout[b * KTOT + DD + tid] = f2bf(r);
}

// ---------------------------------------------------------------------------
// Final prediction: out[b,j] = relu(q_star[b,:] . w_pred[j,:] + b_pred[j])
// ---------------------------------------------------------------------------
__global__ __launch_bounds__(512) void pred_kernel(
    const float* __restrict__ qstar,   // [64][1024]
    const float* __restrict__ w_pred,  // [8][1024]
    const float* __restrict__ b_pred,  // [8]
    float* __restrict__ out)           // [64][8]
{
    int tid = threadIdx.x;   // 512 = 64*8
    int b = tid >> 3;
    int j = tid & 7;
    const float* q = qstar + b * HID;
    const float* wv = w_pred + j * HID;
    float s = b_pred[j];
    for (int k = 0; k < HID; ++k) s += q[k] * wv[k];
    out[tid] = fmaxf(s, 0.f);
}

// ---------------------------------------------------------------------------
// Workspace layout (bytes):
//   [0)                Wbf   bf16 [2048][1536]   6,291,456
//   [6,291,456)        X0    bf16 [64][1536]       196,608
//   [6,488,064)        X1    bf16 [64][1536]       196,608
//   [6,684,672)        c     f32  [64][512]        131,072
//   [6,815,744)        qstar f32  [64][1024]       262,144
//   total                                        7,077,888
// ---------------------------------------------------------------------------
extern "C" void kernel_launch(void* const* d_in, const int* in_sizes, int n_in,
                              void* d_out, int out_size, void* d_ws, size_t ws_size,
                              hipStream_t stream) {
    const float* emb    = (const float*)d_in[0];
    const float* w_ih   = (const float*)d_in[1];
    const float* w_hh   = (const float*)d_in[2];
    const float* b_ih   = (const float*)d_in[3];
    const float* b_hh   = (const float*)d_in[4];
    const float* w_pred = (const float*)d_in[5];
    const float* b_pred = (const float*)d_in[6];

    char* ws = (char*)d_ws;
    unsigned short* Wbf   = (unsigned short*)(ws + 0);
    unsigned short* X0    = (unsigned short*)(ws + 6291456);
    unsigned short* X1    = (unsigned short*)(ws + 6488064);
    float*          cbuf  = (float*)(ws + 6684672);
    float*          qstar = (float*)(ws + 6815744);

    // one-time (per call) weight conversion + state zeroing
    convert_w_kernel<<<(GOUT * KTOT + 255) / 256, 256, 0, stream>>>(w_ih, w_hh, Wbf);
    // zero X0, X1, c, qstar: 786,432 bytes = 196,608 dwords, contiguous from X0
    zero_kernel<<<768, 256, 0, stream>>>((unsigned int*)X0, 196608);

    for (int t = 0; t < NN; ++t) {
        unsigned short* Xin  = (t & 1) ? X1 : X0;
        unsigned short* Xout = (t & 1) ? X0 : X1;
        gates_kernel<<<32, 128, 0, stream>>>(Xin, Wbf, b_ih, b_hh, cbuf, qstar, Xout);
        attn_kernel<<<64, 512, 0, stream>>>(emb, qstar, Xout);
    }

    pred_kernel<<<1, 512, 0, stream>>>(qstar, w_pred, b_pred, (float*)d_out);
}
